// DynamicHead_48120813584686
// MI455X (gfx1250) — matrix operand order repaired and checked by reading, not compiled, not run on hardware
//
#include <hip/hip_runtime.h>

typedef __attribute__((ext_vector_type(16))) _Float16 v16h;
typedef __attribute__((ext_vector_type(8)))  _Float16 v8h;
typedef __attribute__((ext_vector_type(8)))  float    v8f;

#define MID      64
#define NT       128      // pixels per block tile
#define STRIDE   72       // padded half-stride (144 B, 16-B aligned, low bank conflict)
#define HW       4000     // 40*100
#define W_IMG    100
#define H_IMG    40
#define P_STRIDE 8513     // params per instance

// A-matrix (16x32 f16) fragment: lane L holds row M=L&15;
// halves 0-7 = K = kstep*32 + 8*(L>>4) + 0..7 ; halves 8-15 = +16.
__device__ static inline v16h load_frag_A(const _Float16* base, int lane, int kstep) {
    const _Float16* row = base + (lane & 15) * STRIDE;
    int khalf = kstep * 32 + (lane >> 4) * 8;
    union { v16h v; v8h h[2]; } u;
    u.h[0] = *(const v8h*)(row + khalf);
    u.h[1] = *(const v8h*)(row + khalf + 16);
    return u.v;
}

// B-matrix (32x16 f16) fragment: lane L holds col N=L&15;
// halves 0-7 = K = kstep*32 + 8*(L>>4) + 0..7 ; halves 8-15 = +16.
__device__ static inline v16h load_frag_B(const _Float16* base, int lane, int kstep) {
    const _Float16* row = base + (lane & 15) * STRIDE;
    int khalf = kstep * 32 + (lane >> 4) * 8;
    union { v16h v; v8h h[2]; } u;
    u.h[0] = *(const v8h*)(row + khalf);
    u.h[1] = *(const v8h*)(row + khalf + 16);
    return u.v;
}

__global__ __launch_bounds__(256) void dynhead_wmma_kernel(
    const float* __restrict__ x, const float* __restrict__ params,
    float* __restrict__ out)
{
    __shared__ _Float16 sA[NT * STRIDE];   // feat0, later h1
    __shared__ _Float16 sB[NT * STRIDE];   // h0
    __shared__ _Float16 sW0[MID * STRIDE];
    __shared__ _Float16 sW1[MID * STRIDE];
    __shared__ float sB0[MID], sB1[MID], sC0[MID], sC1[MID], sW2[MID];
    __shared__ float sBias2;

    const int tile = blockIdx.x;           // 0..31
    const int n    = blockIdx.y;           // 0..127 instance
    const int tid  = threadIdx.x;
    const int lane = tid & 31;
    const int wave = tid >> 5;
    const int p0   = tile * NT;

    const float* prm = params + (size_t)n * P_STRIDE;
    const float* xb  = x + (size_t)(n >> 4) * ((size_t)MID * HW);

    // ---- cooperative loads: weights (f32 -> f16 LDS), biases, coord weights ----
    for (int idx = tid; idx < MID * MID; idx += 256) {
        int m = idx >> 6, c = idx & 63;
        sW0[m * STRIDE + c] = (_Float16)prm[m * 66 + c + 2];          // skip 2 coord cols
        sW1[m * STRIDE + c] = (_Float16)prm[4224 + m * 64 + c];
    }
    if (tid < MID) {
        sC0[tid] = prm[tid * 66 + 0];
        sC1[tid] = prm[tid * 66 + 1];
        sW2[tid] = prm[8320 + tid];
        sB0[tid] = prm[8384 + tid];
        sB1[tid] = prm[8448 + tid];
    }
    if (tid == 0) sBias2 = prm[8512] - 2.19f;

    // ---- feature tile, transposed into LDS: sA[pixel][channel] ----
    for (int idx = tid; idx < NT * MID; idx += 256) {
        int c  = idx >> 7;        // /NT
        int pl = idx & (NT - 1);
        int p  = p0 + pl;
        float v = (p < HW) ? xb[(size_t)c * HW + p] : 0.0f;
        sA[pl * STRIDE + c] = (_Float16)v;
    }
    __syncthreads();

    const int m16   = (wave & 3) * 16;        // output-row block
    const int ng0   = (wave >> 2) * 4;        // first of 4 column blocks
    const int colN  = lane & 15;
    const int hi    = lane >> 4;
    const int mbase = m16 + hi * 8;           // acc VGPR v <-> row mbase+v

    // =============== Layer 0: h0 = relu(W0[:,2:]*x + (b0 + rank-2 coord bias)) ===============
    {
        v16h a0 = load_frag_A(sW0 + m16 * STRIDE, lane, 0);
        v16h a1 = load_frag_A(sW0 + m16 * STRIDE, lane, 1);
#pragma unroll
        for (int j = 0; j < 4; ++j) {
            int n16 = (ng0 + j) * 16;
            int pl  = n16 + colN;
            int p   = p0 + pl;
            int pc  = (p < HW) ? p : (HW - 1);
            float lx = (float)(pc % W_IMG) * (1.0f / H_IMG);
            float ly = (float)(pc / W_IMG) * (1.0f / H_IMG);
            v8f acc;
#pragma unroll
            for (int v = 0; v < 8; ++v) {
                int m = mbase + v;
                acc[v] = sB0[m] + sC0[m] * lx + sC1[m] * ly;
            }
            v16h bf0 = load_frag_B(sA + n16 * STRIDE, lane, 0);
            v16h bf1 = load_frag_B(sA + n16 * STRIDE, lane, 1);
            acc = __builtin_amdgcn_wmma_f32_16x16x32_f16(false, a0, false, bf0,
                                                         (short)0, acc, false, false);
            acc = __builtin_amdgcn_wmma_f32_16x16x32_f16(false, a1, false, bf1,
                                                         (short)0, acc, false, false);
            v8h hh;
#pragma unroll
            for (int v = 0; v < 8; ++v) hh[v] = (_Float16)fmaxf(acc[v], 0.0f);
            *(v8h*)(sB + pl * STRIDE + mbase) = hh;   // 16-B aligned ds_store_b128
        }
    }
    __syncthreads();

    // =============== Layer 1: h1 = relu(W1*h0 + b1) ===============
    {
        v16h a0 = load_frag_A(sW1 + m16 * STRIDE, lane, 0);
        v16h a1 = load_frag_A(sW1 + m16 * STRIDE, lane, 1);
#pragma unroll
        for (int j = 0; j < 4; ++j) {
            int n16 = (ng0 + j) * 16;
            int pl  = n16 + colN;
            v8f acc;
#pragma unroll
            for (int v = 0; v < 8; ++v) acc[v] = sB1[mbase + v];
            v16h bf0 = load_frag_B(sB + n16 * STRIDE, lane, 0);
            v16h bf1 = load_frag_B(sB + n16 * STRIDE, lane, 1);
            acc = __builtin_amdgcn_wmma_f32_16x16x32_f16(false, a0, false, bf0,
                                                         (short)0, acc, false, false);
            acc = __builtin_amdgcn_wmma_f32_16x16x32_f16(false, a1, false, bf1,
                                                         (short)0, acc, false, false);
            v8h hh;
#pragma unroll
            for (int v = 0; v < 8; ++v) hh[v] = (_Float16)fmaxf(acc[v], 0.0f);
            *(v8h*)(sA + pl * STRIDE + mbase) = hh;   // h1 overwrites feat buffer
        }
    }
    __syncthreads();

    // =============== Layer 2: out = w2 . h1 + (b2 - 2.19) — M=1, do on VALU ===============
    if (tid < NT) {
        int p = p0 + tid;
        if (p < HW) {
            float acc = sBias2;
#pragma unroll
            for (int c8 = 0; c8 < 8; ++c8) {
                v8h hv = *(const v8h*)(sA + tid * STRIDE + c8 * 8);
#pragma unroll
                for (int v = 0; v < 8; ++v)
                    acc += sW2[c8 * 8 + v] * (float)hv[v];
            }
            out[(size_t)n * HW + p] = acc;
        }
    }
}

extern "C" void kernel_launch(void* const* d_in, const int* in_sizes, int n_in,
                              void* d_out, int out_size, void* d_ws, size_t ws_size,
                              hipStream_t stream) {
    const float* x      = (const float*)d_in[0];   // (8,64,40,100) f32
    const float* params = (const float*)d_in[1];   // (128, 8513) f32
    // d_in[2] = num_ins (==16), baked into the kernel's indexing
    float* out = (float*)d_out;                    // (128, 40, 100) f32

    dim3 grid((HW + NT - 1) / NT, 128);            // 32 pixel tiles x 128 instances
    dynhead_wmma_kernel<<<grid, 256, 0, stream>>>(x, params, out);
}
